// MultiHeadStackedGAT_5987184410676
// MI455X (gfx1250) — compile-verified, matrix operand order checked by reading, not run on hardware
//
#include <hip/hip_runtime.h>
#include <math.h>

// ---------------------------------------------------------------------------
// MultiHeadStackedGAT forward for MI455X (gfx1250, wave32, WMMA f16)
//   N=25000, E=400000, H=64, NH=4, HS=16, S=16, EF=8, K=2 hops
// Dense contractions -> v_wmma_f32_16x16x32_f16 (16-row tiles per wave32)
// Irregular segment ops -> float global atomics (native on CDNA5)
// ---------------------------------------------------------------------------

typedef __attribute__((ext_vector_type(16))) _Float16 v16h;
typedef __attribute__((ext_vector_type(8)))  float    v8f;

#define WMMA_F16(a, b, c) \
  __builtin_amdgcn_wmma_f32_16x16x32_f16(false, (a), false, (b), (short)0, (c), false, false)

// K index held by half j (0..15) of an A/B fragment, per CDNA5 ISA 7.12.2
__device__ __forceinline__ int frag_k(int j, int lane) {
  return ((j >> 3) << 4) + ((lane & 16) ? 8 : 0) + (j & 7);
}

__device__ __forceinline__ void atomicMaxFloat(float* addr, float val) {
  if (val >= 0.0f) atomicMax((int*)addr, __float_as_int(val));
  else             atomicMin((unsigned int*)addr, __float_as_uint(val));
}

// ------------------------------- weight converts ---------------------------

__global__ void cvt_f16_kernel(const float* __restrict__ in, _Float16* __restrict__ out, int n) {
  int i = blockIdx.x * blockDim.x + threadIdx.x;
  if (i < n) out[i] = (_Float16)in[i];
}

// w1 [128][168] f32 -> [128][192] f16 (K padded with zeros)
__global__ void cvt_w1_kernel(const float* __restrict__ in, _Float16* __restrict__ out) {
  int i = blockIdx.x * blockDim.x + threadIdx.x;
  if (i >= 128 * 192) return;
  int o = i / 192, k = i - o * 192;
  out[i] = (_Float16)(k < 168 ? in[o * 168 + k] : 0.0f);
}

// w2 [4][128] f32 -> B2 [128][16] f16 (cols 4..15 zero): B2[k][n] = w2[n][k]
__global__ void cvt_w2pad_kernel(const float* __restrict__ in, _Float16* __restrict__ out) {
  int i = blockIdx.x * blockDim.x + threadIdx.x;
  if (i >= 128 * 16) return;
  int k = i >> 4, n = i & 15;
  out[i] = (_Float16)(n < 4 ? in[n * 128 + k] : 0.0f);
}

// ------------------------------- small VALU kernels ------------------------

__global__ void fill_kernel(float* __restrict__ p, float v, long long n) {
  long long i = (long long)blockIdx.x * blockDim.x + threadIdx.x;
  if (i < n) p[i] = v;
}

// LayerNorm over H=64; 64 threads per row, 4 rows per 256-thread block
__global__ void ln_kernel(const float* __restrict__ h, const float* __restrict__ lnw,
                          const float* __restrict__ lnb, float* __restrict__ out, int n) {
  __shared__ float sbuf[4][64];
  int g = threadIdx.x >> 6, c = threadIdx.x & 63;
  int row = blockIdx.x * 4 + g;
  bool ok = row < n;
  float v = ok ? h[(long long)row * 64 + c] : 0.0f;
  sbuf[g][c] = v;
  __syncthreads();
  for (int s = 32; s > 0; s >>= 1) { if (c < s) sbuf[g][c] += sbuf[g][c + s]; __syncthreads(); }
  float mean = sbuf[g][0] * (1.0f / 64.0f);
  __syncthreads();
  float d = v - mean;
  sbuf[g][c] = d * d;
  __syncthreads();
  for (int s = 32; s > 0; s >>= 1) { if (c < s) sbuf[g][c] += sbuf[g][c + s]; __syncthreads(); }
  float var = sbuf[g][0] * (1.0f / 64.0f);
  if (ok) out[(long long)row * 64 + c] = d * rsqrtf(var + 1e-5f) * lnw[c] + lnb[c];
}

__device__ __forceinline__ float lrelu_scaled(float s) {
  return (s > 0.0f ? s : 0.01f * s) * 0.25f;  // leaky_relu / sqrt(HS=16)
}

__global__ void seg_max_kernel(const float* __restrict__ raw, const int* __restrict__ dst,
                               float* __restrict__ segmax, int E_) {
  int i = blockIdx.x * blockDim.x + threadIdx.x;
  if (i >= E_ * 4) return;
  int e = i >> 2, hd = i & 3;
  atomicMaxFloat(&segmax[(long long)dst[e] * 4 + hd], lrelu_scaled(raw[i]));
}

__global__ void seg_exp_kernel(const float* __restrict__ raw, const int* __restrict__ dst,
                               const float* __restrict__ segmax, float* __restrict__ segsum,
                               float* __restrict__ wout, int hop, int E_) {
  int i = blockIdx.x * blockDim.x + threadIdx.x;
  if (i >= E_ * 4) return;
  int e = i >> 2, hd = i & 3;
  float ex = expf(lrelu_scaled(raw[i]) - segmax[(long long)dst[e] * 4 + hd]);
  atomicAdd(&segsum[(long long)dst[e] * 4 + hd], ex);
  wout[(long long)e * 8 + hd * 2 + hop] = ex;   // [E,NH,K] layout of output ws
}

__global__ void seg_norm_kernel(float* __restrict__ wout, const float* __restrict__ segsum,
                                const int* __restrict__ dst, int hop, int E_) {
  int i = blockIdx.x * blockDim.x + threadIdx.x;
  if (i >= E_ * 4) return;
  int e = i >> 2, hd = i & 3;
  long long idx = (long long)e * 8 + hd * 2 + hop;
  wout[idx] = wout[idx] / (segsum[(long long)dst[e] * 4 + hd] + 1e-9f);
}

__global__ void gate_sigmoid_kernel(const float* __restrict__ raw, float* __restrict__ wout,
                                    int hop, int E_) {
  int i = blockIdx.x * blockDim.x + threadIdx.x;
  if (i >= E_ * 4) return;
  int e = i >> 2, hd = i & 3;
  wout[(long long)e * 8 + hd * 2 + hop] = 1.0f / (1.0f + expf(-raw[i]));
}

// agg[dst][f] += h_norm[src][f] * w[e][f/16]   (E*64 threads, f32 atomics)
__global__ void msg_accum_kernel(const float* __restrict__ hnorm, const int* __restrict__ srcp,
                                 const int* __restrict__ dstp, const float* __restrict__ wout,
                                 int hop, float* __restrict__ agg, int E_) {
  long long i = (long long)blockIdx.x * blockDim.x + threadIdx.x;
  if (i >= (long long)E_ * 64) return;
  int e = (int)(i >> 6), f = (int)(i & 63);
  float w = wout[(long long)e * 8 + (f >> 4) * 2 + hop];
  atomicAdd(&agg[(long long)dstp[e] * 64 + f], hnorm[(long long)srcp[e] * 64 + f] * w);
}

__global__ void final_combine_kernel(const float* __restrict__ x, const float* __restrict__ z,
                                     const float* __restrict__ cand, float* __restrict__ out,
                                     long long n) {
  long long i = (long long)blockIdx.x * blockDim.x + threadIdx.x;
  if (i < n) out[i] = (1.0f - z[i]) * x[i] + z[i] * cand[i];
}

// ------------------------------- WMMA kernels ------------------------------

// Per-edge fused MLP: raw[e][h] = relu(inp @ w1^T) @ w2^T
// inp = [h_norm[src] | h_norm[dst] | x_s[src] | x_s[dst] | ef]  (168 -> pad 192)
// One 16-edge tile per wave; 4 waves per block.
__global__ void __launch_bounds__(128)
edge_mlp_kernel(const float* __restrict__ hnorm, const float* __restrict__ xs,
                const float* __restrict__ ef, const int* __restrict__ srcp,
                const int* __restrict__ dstp, const _Float16* __restrict__ w1h,
                const _Float16* __restrict__ b2h, float* __restrict__ raw, int E_) {
  __shared__ _Float16 s_inp[4][16][200];  // 16 edges x 192 features (padded stride)
  __shared__ _Float16 s_hid[4][16][136];  // 16 edges x 128 hidden
  int wave = threadIdx.x >> 5, lane = threadIdx.x & 31;
  int tile = blockIdx.x * 4 + wave;
  long long ebase = (long long)tile * 16;

  __builtin_prefetch(&srcp[ebase], 0, 0);  // global_prefetch_b8
  __builtin_prefetch(&dstp[ebase], 0, 0);

  // stage the 16 x 168 gathered inputs into LDS as f16
  for (int t = lane; t < 16 * 168; t += 32) {
    int m = t / 168, f = t - m * 168;
    long long e = ebase + m;
    float v = 0.0f;
    if (e < E_) {
      long long s = srcp[e], d = dstp[e];
      if (f < 64)       v = hnorm[s * 64 + f];
      else if (f < 128) v = hnorm[d * 64 + (f - 64)];
      else if (f < 144) v = xs[s * 16 + (f - 128)];
      else if (f < 160) v = xs[d * 16 + (f - 144)];
      else              v = ef[e * 8 + (f - 160)];
    }
    s_inp[wave][m][f] = (_Float16)v;
  }
  for (int t = lane; t < 16 * 24; t += 32)
    s_inp[wave][t / 24][168 + t % 24] = (_Float16)0.0f;
  __syncthreads();

  int row = lane & 15;
  int mh = (lane & 16) ? 8 : 0;

  v16h a[6];
#pragma unroll
  for (int kt = 0; kt < 6; ++kt)
#pragma unroll
    for (int j = 0; j < 16; ++j)
      a[kt][j] = s_inp[wave][row][kt * 32 + frag_k(j, lane)];

  // layer 1: [16x192] x [192x128] -> relu -> LDS
#pragma unroll
  for (int nt = 0; nt < 8; ++nt) {
    v8f acc = {};
#pragma unroll
    for (int kt = 0; kt < 6; ++kt) {
      v16h b;
#pragma unroll
      for (int j = 0; j < 16; ++j)
        b[j] = w1h[(nt * 16 + row) * 192 + kt * 32 + frag_k(j, lane)];
      acc = WMMA_F16(a[kt], b, acc);
    }
#pragma unroll
    for (int v = 0; v < 8; ++v) {
      float r = acc[v];
      s_hid[wave][v + mh][nt * 16 + row] = (_Float16)(r > 0.0f ? r : 0.0f);
    }
  }
  __syncthreads();

  // layer 2: [16x128] x [128x16] (cols 0..3 = heads)
  v8f sc = {};
#pragma unroll
  for (int kt = 0; kt < 4; ++kt) {
    v16h a2, b;
#pragma unroll
    for (int j = 0; j < 16; ++j) {
      int k = kt * 32 + frag_k(j, lane);
      a2[j] = s_hid[wave][row][k];
      b[j]  = b2h[k * 16 + row];
    }
    sc = WMMA_F16(a2, b, sc);
  }
  if (row < 4) {
#pragma unroll
    for (int v = 0; v < 8; ++v) {
      long long e = ebase + v + mh;
      if (e < E_) raw[e * 4 + row] = sc[v];
    }
  }
}

// h[n] += agg[n] @ proj^T   ([N x 64] x [64 x 64])
__global__ void __launch_bounds__(128)
node_update_kernel(const float* __restrict__ agg, const _Float16* __restrict__ projh,
                   float* __restrict__ h, int n) {
  int wave = threadIdx.x >> 5, lane = threadIdx.x & 31;
  int tile = blockIdx.x * 4 + wave;
  int rbase = tile * 16;
  int row = lane & 15, mh = (lane & 16) ? 8 : 0;
  int r = rbase + row;
  bool rok = r < n;

  v16h a[2];
#pragma unroll
  for (int kt = 0; kt < 2; ++kt)
#pragma unroll
    for (int j = 0; j < 16; ++j)
      a[kt][j] = (_Float16)(rok ? agg[(long long)r * 64 + kt * 32 + frag_k(j, lane)] : 0.0f);

#pragma unroll
  for (int nt = 0; nt < 4; ++nt) {
    v8f acc = {};
#pragma unroll
    for (int kt = 0; kt < 2; ++kt) {
      v16h b;
#pragma unroll
      for (int j = 0; j < 16; ++j)
        b[j] = projh[(nt * 16 + row) * 64 + kt * 32 + frag_k(j, lane)];
      acc = WMMA_F16(a[kt], b, acc);
    }
#pragma unroll
    for (int v = 0; v < 8; ++v) {
      int rr = rbase + v + mh;
      if (rr < n) h[(long long)rr * 64 + nt * 16 + row] += acc[v];
    }
  }
}

// Fused GRU gate MLP: act((relu(gi @ W1^T + b1)) @ W2^T + b2)
// gi built on the fly: [x | h_fwd - x | h_rev - x], mode 2 uses r*x in cols 0..63.
// mode 0/1 -> sigmoid, mode 2 -> tanh
__global__ void __launch_bounds__(128)
gate_mlp_kernel(const float* __restrict__ x, const float* __restrict__ hf,
                const float* __restrict__ hr, const float* __restrict__ rg,
                const _Float16* __restrict__ w1h, const float* __restrict__ b1,
                const _Float16* __restrict__ w2h, const float* __restrict__ b2,
                float* __restrict__ out, int mode, int n) {
  __shared__ _Float16 s_hid[4][16][200];  // 16 rows x 192 hidden
  int wave = threadIdx.x >> 5, lane = threadIdx.x & 31;
  int tile = blockIdx.x * 4 + wave;
  int rbase = tile * 16;
  int row = lane & 15, mh = (lane & 16) ? 8 : 0;
  int r = rbase + row;
  bool rok = r < n;

  v16h a[6];
#pragma unroll
  for (int kt = 0; kt < 6; ++kt)
#pragma unroll
    for (int j = 0; j < 16; ++j) {
      int k = kt * 32 + frag_k(j, lane);
      float v = 0.0f;
      if (rok) {
        long long base = (long long)r * 64;
        if (k < 64)       v = (mode == 2) ? rg[base + k] * x[base + k] : x[base + k];
        else if (k < 128) v = hf[base + (k - 64)] - x[base + (k - 64)];
        else              v = hr[base + (k - 128)] - x[base + (k - 128)];
      }
      a[kt][j] = (_Float16)v;
    }

  // layer 1: [16x192] x [192x192] + b1 -> relu -> LDS
#pragma unroll
  for (int nt = 0; nt < 12; ++nt) {
    v8f acc = {};
#pragma unroll
    for (int kt = 0; kt < 6; ++kt) {
      v16h b;
#pragma unroll
      for (int j = 0; j < 16; ++j)
        b[j] = w1h[(nt * 16 + row) * 192 + kt * 32 + frag_k(j, lane)];
      acc = WMMA_F16(a[kt], b, acc);
    }
    float bias = b1[nt * 16 + row];
#pragma unroll
    for (int v = 0; v < 8; ++v) {
      float t = acc[v] + bias;
      s_hid[wave][v + mh][nt * 16 + row] = (_Float16)(t > 0.0f ? t : 0.0f);
    }
  }
  __syncthreads();

  // layer 2: [16x192] x [192x64] + b2 -> activation
  v16h a2[6];
#pragma unroll
  for (int kt = 0; kt < 6; ++kt)
#pragma unroll
    for (int j = 0; j < 16; ++j)
      a2[kt][j] = s_hid[wave][row][kt * 32 + frag_k(j, lane)];

#pragma unroll
  for (int nt = 0; nt < 4; ++nt) {
    v8f acc = {};
#pragma unroll
    for (int kt = 0; kt < 6; ++kt) {
      v16h b;
#pragma unroll
      for (int j = 0; j < 16; ++j)
        b[j] = w2h[(nt * 16 + row) * 192 + kt * 32 + frag_k(j, lane)];
      acc = WMMA_F16(a2[kt], b, acc);
    }
    float bias = b2[nt * 16 + row];
#pragma unroll
    for (int v = 0; v < 8; ++v) {
      int rr = rbase + v + mh;
      if (rr < n) {
        float t = acc[v] + bias;
        out[(long long)rr * 64 + nt * 16 + row] =
            (mode == 2) ? tanhf(t) : 1.0f / (1.0f + expf(-t));
      }
    }
  }
}

// ------------------------------- host driver -------------------------------

extern "C" void kernel_launch(void* const* d_in, const int* in_sizes, int n_in,
                              void* d_out, int out_size, void* d_ws, size_t ws_size,
                              hipStream_t stream) {
  const float* x      = (const float*)d_in[0];
  const float* xs     = (const float*)d_in[1];
  const int*   ei     = (const int*)d_in[2];      // [2][E]
  const float* ef     = (const float*)d_in[3];    // [E][8]
  const float* fwd_w1 = (const float*)d_in[4];    // [2][128][168]
  const float* fwd_w2 = (const float*)d_in[5];    // [2][4][128]
  const float* fwd_pj = (const float*)d_in[6];    // [2][64][64]
  const float* rev_w1 = (const float*)d_in[7];
  const float* rev_w2 = (const float*)d_in[8];
  const float* rev_pj = (const float*)d_in[9];
  const float* lnw    = (const float*)d_in[10];
  const float* lnb    = (const float*)d_in[11];
  const float* gw1[3] = { (const float*)d_in[12], (const float*)d_in[16], (const float*)d_in[20] };
  const float* gb1[3] = { (const float*)d_in[13], (const float*)d_in[17], (const float*)d_in[21] };
  const float* gw2[3] = { (const float*)d_in[14], (const float*)d_in[18], (const float*)d_in[22] };
  const float* gb2[3] = { (const float*)d_in[15], (const float*)d_in[19], (const float*)d_in[23] };

  const int N = in_sizes[0] / 64;
  const int E = in_sizes[2] / 2;
  const long long NH64 = (long long)N * 64;

  // output slices (return order: final_h, fwd_ws, rev_ws, z_gate, r_gate)
  float* out_final = (float*)d_out;
  float* out_fwdws = out_final + NH64;
  float* out_revws = out_fwdws + (long long)E * 8;
  float* out_z     = out_revws + (long long)E * 8;
  float* out_r     = out_z + NH64;

  // workspace carve
  size_t off = 0;
  auto carve = [&](size_t bytes) {
    void* p = (char*)d_ws + off;
    off = (off + bytes + 255) & ~(size_t)255;
    return p;
  };
  float* h_fwd  = (float*)carve(NH64 * 4);
  float* h_rev  = (float*)carve(NH64 * 4);
  float* h_norm = (float*)carve(NH64 * 4);
  float* agg    = (float*)carve(NH64 * 4);           // reused as cand at the end
  float* raw    = (float*)carve((size_t)E * 4 * 4);
  float* segmax = (float*)carve((size_t)N * 4 * 4);
  float* segsum = (float*)carve((size_t)N * 4 * 4);
  _Float16* w1h_all  = (_Float16*)carve((size_t)4 * 128 * 192 * 2);
  _Float16* b2h_all  = (_Float16*)carve((size_t)4 * 128 * 16 * 2);
  _Float16* pjh_all  = (_Float16*)carve((size_t)4 * 64 * 64 * 2);
  _Float16* gw1h_all = (_Float16*)carve((size_t)3 * 192 * 192 * 2);
  _Float16* gw2h_all = (_Float16*)carve((size_t)3 * 64 * 192 * 2);
  (void)ws_size; (void)n_in; (void)out_size;

  const int T = 256;
  // ---- convert all weights to f16 once
  for (int d = 0; d < 2; ++d)
    for (int k = 0; k < 2; ++k) {
      int idx = d * 2 + k;
      const float* w1s = (d == 0 ? fwd_w1 : rev_w1) + (size_t)k * 128 * 168;
      const float* w2s = (d == 0 ? fwd_w2 : rev_w2) + (size_t)k * 4 * 128;
      const float* pjs = (d == 0 ? fwd_pj : rev_pj) + (size_t)k * 64 * 64;
      cvt_w1_kernel<<<(128 * 192 + T - 1) / T, T, 0, stream>>>(w1s, w1h_all + (size_t)idx * 128 * 192);
      cvt_w2pad_kernel<<<(128 * 16 + T - 1) / T, T, 0, stream>>>(w2s, b2h_all + (size_t)idx * 128 * 16);
      cvt_f16_kernel<<<(64 * 64 + T - 1) / T, T, 0, stream>>>(pjs, pjh_all + (size_t)idx * 64 * 64, 64 * 64);
    }
  for (int g = 0; g < 3; ++g) {
    cvt_f16_kernel<<<(192 * 192 + T - 1) / T, T, 0, stream>>>(gw1[g], gw1h_all + (size_t)g * 192 * 192, 192 * 192);
    cvt_f16_kernel<<<(64 * 192 + T - 1) / T, T, 0, stream>>>(gw2[g], gw2h_all + (size_t)g * 64 * 192, 64 * 192);
  }

  hipMemcpyAsync(h_fwd, x, NH64 * 4, hipMemcpyDeviceToDevice, stream);
  hipMemcpyAsync(h_rev, x, NH64 * 4, hipMemcpyDeviceToDevice, stream);

  const int lnBlocks   = (N + 3) / 4;
  const int edgeTiles  = (E + 15) / 16;
  const int edgeBlocks = (edgeTiles + 3) / 4;
  const int e4Blocks   = (E * 4 + T - 1) / T;
  const int e64Blocks  = (int)(((long long)E * 64 + T - 1) / T);
  const int nodeTiles  = (N + 15) / 16;
  const int nodeBlocks = (nodeTiles + 3) / 4;
  const int nfBlocks   = (int)((NH64 + T - 1) / T);

  const int* eSrcF = ei;       // fwd: src=ei[0], dst=ei[1]
  const int* eDstF = ei + E;
  const int* eSrcR = ei + E;   // rev: flipped
  const int* eDstR = ei;

  // ---- forward hops (segment softmax attention)
  for (int k = 0; k < 2; ++k) {
    int idx = k;  // dir 0
    ln_kernel<<<lnBlocks, 256, 0, stream>>>(h_fwd, lnw, lnb, h_norm, N);
    edge_mlp_kernel<<<edgeBlocks, 128, 0, stream>>>(h_norm, xs, ef, eSrcF, eDstF,
        w1h_all + (size_t)idx * 128 * 192, b2h_all + (size_t)idx * 128 * 16, raw, E);
    fill_kernel<<<(N * 4 + T - 1) / T, T, 0, stream>>>(segmax, -__builtin_inff(), (long long)N * 4);
    fill_kernel<<<(N * 4 + T - 1) / T, T, 0, stream>>>(segsum, 0.0f, (long long)N * 4);
    fill_kernel<<<nfBlocks, T, 0, stream>>>(agg, 0.0f, NH64);
    seg_max_kernel<<<e4Blocks, T, 0, stream>>>(raw, eDstF, segmax, E);
    seg_exp_kernel<<<e4Blocks, T, 0, stream>>>(raw, eDstF, segmax, segsum, out_fwdws, k, E);
    seg_norm_kernel<<<e4Blocks, T, 0, stream>>>(out_fwdws, segsum, eDstF, k, E);
    msg_accum_kernel<<<e64Blocks, T, 0, stream>>>(h_norm, eSrcF, eDstF, out_fwdws, k, agg, E);
    node_update_kernel<<<nodeBlocks, 128, 0, stream>>>(agg, pjh_all + (size_t)idx * 64 * 64, h_fwd, N);
  }

  // ---- reverse hops (sigmoid gating, edge direction flipped)
  for (int k = 0; k < 2; ++k) {
    int idx = 2 + k;  // dir 1
    ln_kernel<<<lnBlocks, 256, 0, stream>>>(h_rev, lnw, lnb, h_norm, N);
    edge_mlp_kernel<<<edgeBlocks, 128, 0, stream>>>(h_norm, xs, ef, eSrcR, eDstR,
        w1h_all + (size_t)idx * 128 * 192, b2h_all + (size_t)idx * 128 * 16, raw, E);
    fill_kernel<<<nfBlocks, T, 0, stream>>>(agg, 0.0f, NH64);
    gate_sigmoid_kernel<<<e4Blocks, T, 0, stream>>>(raw, out_revws, k, E);
    msg_accum_kernel<<<e64Blocks, T, 0, stream>>>(h_norm, eSrcR, eDstR, out_revws, k, agg, E);
    node_update_kernel<<<nodeBlocks, 128, 0, stream>>>(agg, pjh_all + (size_t)idx * 64 * 64, h_rev, N);
  }

  // ---- GRU update
  gate_mlp_kernel<<<nodeBlocks, 128, 0, stream>>>(x, h_fwd, h_rev, out_r,
      gw1h_all + 0 * 192 * 192, gb1[0], gw2h_all + 0 * 64 * 192, gb2[0], out_r, 0, N);
  gate_mlp_kernel<<<nodeBlocks, 128, 0, stream>>>(x, h_fwd, h_rev, out_r,
      gw1h_all + 1 * 192 * 192, gb1[1], gw2h_all + 1 * 64 * 192, gb2[1], out_z, 1, N);
  gate_mlp_kernel<<<nodeBlocks, 128, 0, stream>>>(x, h_fwd, h_rev, out_r,
      gw1h_all + 2 * 192 * 192, gb1[2], gw2h_all + 2 * 64 * 192, gb2[2], agg /*cand*/, 2, N);
  final_combine_kernel<<<nfBlocks, T, 0, stream>>>(x, out_z, agg, out_final, NH64);
}
